// ActionDecoderCrossAttention_75685913690647
// MI455X (gfx1250) — matrix operand order repaired and checked by reading, not compiled, not run on hardware
//
#include <hip/hip_runtime.h>

// ---------------------------------------------------------------------------
// Types for CDNA5 WMMA (wave32): v_wmma_f32_16x16x32_bf16
// ---------------------------------------------------------------------------
typedef __bf16 bf16_t;
typedef __attribute__((ext_vector_type(16))) __bf16 v16bf;
typedef __attribute__((ext_vector_type(8)))  __bf16 v8bf;
typedef __attribute__((ext_vector_type(8)))  float  v8f;

// Problem constants (reference shapes)
#define BATCH 2
#define LSEQ  1024
#define SSEQ  4096
#define DMODEL 1024
#define CAD   1024
#define NHEADS 16
#define DHEAD 64

// ---------------------------------------------------------------------------
// A-fragment loader: 16-bit A 16x32, this code's own k order (not the ISA table order)
//   lane r = lane%16 holds row M=r; half = lane/16
//   element t: K = half*16 + t (contiguous)
// ---------------------------------------------------------------------------
__device__ inline v16bf load_a_frag(const bf16_t* rowkk, int half) {
    v8bf lo = *reinterpret_cast<const v8bf*>(rowkk + half * 16);
    v8bf hi = *reinterpret_cast<const v8bf*>(rowkk + half * 16 + 8);
    v16bf r;
#pragma unroll
    for (int i = 0; i < 8; ++i) { r[i] = lo[i]; r[i + 8] = hi[i]; }
    return r;
}

// B-fragment: 16-bit B 32x16 layout — lane (half, c): column n=c,
// element t: K = half*16 + t (contiguous). 32B contiguous, 16B-aligned.
__device__ inline v16bf load_b_frag(const bf16_t* p) {
    v8bf lo = *reinterpret_cast<const v8bf*>(p);
    v8bf hi = *reinterpret_cast<const v8bf*>(p + 8);
    v16bf r;
#pragma unroll
    for (int i = 0; i < 8; ++i) { r[i] = lo[i]; r[i + 8] = hi[i]; }
    return r;
}

// ---------------------------------------------------------------------------
// RMSNorm (x = h / (||h||/sqrt(D) + eps) * scale) -> bf16
// ---------------------------------------------------------------------------
__global__ __launch_bounds__(256)
void rmsnorm_kernel(const float* __restrict__ hs, const float* __restrict__ scale,
                    bf16_t* __restrict__ xb) {
    const int row = blockIdx.x;
    const float* hp = hs + (size_t)row * DMODEL;
    float ss = 0.f;
#pragma unroll
    for (int i = 0; i < 4; ++i) {
        float v = hp[threadIdx.x + i * 256];
        ss += v * v;
    }
#pragma unroll
    for (int d = 1; d < 32; d <<= 1) ss += __shfl_xor(ss, d, 32);
    __shared__ float red[8];
    if ((threadIdx.x & 31) == 0) red[threadIdx.x >> 5] = ss;
    __syncthreads();
    float tot = 0.f;
#pragma unroll
    for (int w = 0; w < 8; ++w) tot += red[w];
    float rms = sqrtf(tot) * 0.03125f;   // * D^-0.5, D=1024
    float inv = 1.0f / (rms + 1e-8f);
#pragma unroll
    for (int i = 0; i < 4; ++i) {
        int d = threadIdx.x + i * 256;
        xb[(size_t)row * DMODEL + d] = (bf16_t)(hp[d] * inv * scale[d]);
    }
}

// f32 -> bf16 elementwise
__global__ void cvt_bf16_kernel(const float* __restrict__ src,
                                bf16_t* __restrict__ dst, int n) {
    int i = blockIdx.x * blockDim.x + threadIdx.x;
    if (i < n) dst[i] = (bf16_t)src[i];
}

// Wt[n*K + k] = (bf16) W[k*N + n]   (32x32 LDS tile, 256 thr = 32x8)
__global__ __launch_bounds__(256)
void transpose_cvt_kernel(const float* __restrict__ W, bf16_t* __restrict__ Wt,
                          int K, int N) {
    __shared__ float tile[32][33];
    int n0 = blockIdx.x * 32, k0 = blockIdx.y * 32;
    int tx = threadIdx.x & 31, ty = threadIdx.x >> 5;
#pragma unroll
    for (int i = 0; i < 32; i += 8)
        tile[ty + i][tx] = W[(size_t)(k0 + ty + i) * N + n0 + tx];
    __syncthreads();
#pragma unroll
    for (int i = 0; i < 32; i += 8)
        Wt[(size_t)(n0 + ty + i) * K + k0 + tx] = (bf16_t)tile[tx][ty + i];
}

// ---------------------------------------------------------------------------
// WMMA GEMM:  C[M,N] = A[M,K] (bf16 row-major) x Bt (bf16, Bt[n*K+k])
// One wave per 32x64 output tile: 2 A-frags x 4 B-frags -> 8 WMMA per K-step.
// ---------------------------------------------------------------------------
template <bool F32OUT>
__global__ __launch_bounds__(256)
void gemm_wmma_kernel(const bf16_t* __restrict__ A, const bf16_t* __restrict__ Bt,
                      bf16_t* __restrict__ Cb, float* __restrict__ Cf,
                      int M, int N, int K) {
    int wid = blockIdx.x * (blockDim.x >> 5) + (threadIdx.x >> 5);
    int tiles_m = M >> 5, tiles_n = N >> 6;
    if (wid >= tiles_m * tiles_n) return;
    int tn = wid / tiles_m, tm = wid % tiles_m;
    int lane = threadIdx.x & 31;
    int half = lane >> 4, lc = lane & 15;

    const bf16_t* arow0 = A + (size_t)((tm << 5) + lc) * K;
    const bf16_t* arow1 = A + (size_t)((tm << 5) + 16 + lc) * K;

    v8f acc[2][4];
#pragma unroll
    for (int mi = 0; mi < 2; ++mi)
#pragma unroll
        for (int j = 0; j < 4; ++j)
#pragma unroll
            for (int i = 0; i < 8; ++i) acc[mi][j][i] = 0.f;

    for (int kk = 0; kk < K; kk += 32) {
        v16bf a0 = load_a_frag(arow0 + kk, half);
        v16bf a1 = load_a_frag(arow1 + kk, half);
#pragma unroll
        for (int j = 0; j < 4; ++j) {
            int n = (tn << 6) + (j << 4) + lc;
            v16bf b = load_b_frag(Bt + (size_t)n * K + kk + half * 16);
            acc[0][j] = __builtin_amdgcn_wmma_f32_16x16x32_bf16(
                false, a0, false, b, (short)0, acc[0][j], false, false);
            acc[1][j] = __builtin_amdgcn_wmma_f32_16x16x32_bf16(
                false, a1, false, b, (short)0, acc[1][j], false, false);
        }
    }
#pragma unroll
    for (int mi = 0; mi < 2; ++mi)
#pragma unroll
        for (int j = 0; j < 4; ++j) {
            int n = (tn << 6) + (j << 4) + lc;
#pragma unroll
            for (int i = 0; i < 8; ++i) {
                int m = (tm << 5) + (mi << 4) + i + half * 8;
                if (F32OUT) Cf[(size_t)m * N + n] = acc[mi][j][i];
                else        Cb[(size_t)m * N + n] = (bf16_t)acc[mi][j][i];
            }
        }
}

// ---------------------------------------------------------------------------
// Flash attention (no 1/sqrt(dh) scaling, faithful to reference):
//   grid (B*h, L/128), block 256 (8 waves of 16 q-rows each).
//   64 S-columns per iteration: 8 WMMA for QK^T + 8 WMMA for PV, one softmax
//   butterfly per row per 64 columns.
//   q: [B*L, CA] bf16 ; kv: [B*S, 2*CA] bf16 (k at +0, v at +CA)
// ---------------------------------------------------------------------------
__global__ __launch_bounds__(256)
void attn_kernel(const bf16_t* __restrict__ qb, const bf16_t* __restrict__ kvb,
                 const float* __restrict__ maskp, bf16_t* __restrict__ attnb) {
    const int b    = blockIdx.x >> 4;
    const int head = blockIdx.x & 15;
    const int wave = threadIdx.x >> 5;
    const int lane = threadIdx.x & 31;
    const int half = lane >> 4, lc = lane & 15;
    const int l0 = blockIdx.y * 128 + wave * 16;

    __shared__ __align__(16) bf16_t sK[64][72];     // K tile row-major (144B rows)
    __shared__ __align__(16) bf16_t sVt[64][72];    // V tile transposed: sVt[d][s_local]
    __shared__ __align__(16) bf16_t sP[8][16][64];  // per-wave probs (16x64)

    // q fragments (dh=64 -> two K-steps of 32), registers for the whole S loop
    const bf16_t* qrow = qb + ((size_t)(b * LSEQ + l0 + lc)) * CAD + head * DHEAD;
    v16bf qa0 = load_a_frag(qrow, half);
    v16bf qa1 = load_a_frag(qrow + 32, half);

    float m_i[8], l_i[8];
    v8f acc[4];
#pragma unroll
    for (int i = 0; i < 8; ++i) { m_i[i] = -3.0e38f; l_i[i] = 0.f; }
#pragma unroll
    for (int j = 0; j < 4; ++j)
#pragma unroll
        for (int i = 0; i < 8; ++i) acc[j][i] = 0.f;

    const bf16_t* kbase = kvb + ((size_t)b * SSEQ) * (2 * CAD) + head * DHEAD;

    for (int s0 = 0; s0 < SSEQ; s0 += 64) {
        // Cooperative stage: 64 rows x 64 bf16 of K and V (V transposed).
        {
            int r_  = threadIdx.x >> 2;   // 0..63
            int seg = threadIdx.x & 3;    // 0..3  (16 bf16 = 32B per thread)
            const bf16_t* kp = kbase + (size_t)(s0 + r_) * (2 * CAD) + seg * 16;
            v8bf k0 = *reinterpret_cast<const v8bf*>(kp);
            v8bf k1 = *reinterpret_cast<const v8bf*>(kp + 8);
            *reinterpret_cast<v8bf*>(&sK[r_][seg * 16])     = k0;
            *reinterpret_cast<v8bf*>(&sK[r_][seg * 16 + 8]) = k1;
            v8bf v0 = *reinterpret_cast<const v8bf*>(kp + CAD);
            v8bf v1 = *reinterpret_cast<const v8bf*>(kp + CAD + 8);
#pragma unroll
            for (int e = 0; e < 8; ++e) {
                sVt[seg * 16 + e][r_]     = v0[e];
                sVt[seg * 16 + 8 + e][r_] = v1[e];
            }
            // prefetch next tile while this step computes
            if (s0 + 64 < SSEQ)
                __builtin_prefetch((const void*)(kp + (size_t)64 * 2 * CAD), 0, 0);
        }
        __syncthreads();

        // scores[16 x 64] = q(16x64) @ K^T : 8 WMMA
        v8f sc[4];
#pragma unroll
        for (int j = 0; j < 4; ++j)
#pragma unroll
            for (int i = 0; i < 8; ++i) sc[j][i] = 0.f;
#pragma unroll
        for (int j = 0; j < 4; ++j) {
            v16bf bk0 = load_b_frag(&sK[j * 16 + lc][half * 16]);
            sc[j] = __builtin_amdgcn_wmma_f32_16x16x32_bf16(
                false, qa0, false, bk0, (short)0, sc[j], false, false);
            v16bf bk1 = load_b_frag(&sK[j * 16 + lc][32 + half * 16]);
            sc[j] = __builtin_amdgcn_wmma_f32_16x16x32_bf16(
                false, qa1, false, bk1, (short)0, sc[j], false, false);
        }
        float madd[4];
#pragma unroll
        for (int j = 0; j < 4; ++j)
            madd[j] = (1.0f - maskp[b * SSEQ + s0 + j * 16 + lc]) * -10000.0f;

        // Online softmax. Row M = i + half*8 lives in the 16 lanes of this
        // half; shfl_xor masks 1..8 reduce within the half.
#pragma unroll
        for (int i = 0; i < 8; ++i) {
            float v0 = sc[0][i] + madd[0];
            float v1 = sc[1][i] + madd[1];
            float v2 = sc[2][i] + madd[2];
            float v3 = sc[3][i] + madd[3];
            float mx = fmaxf(fmaxf(v0, v1), fmaxf(v2, v3));
#pragma unroll
            for (int d = 1; d < 16; d <<= 1) mx = fmaxf(mx, __shfl_xor(mx, d, 32));
            float nm = fmaxf(m_i[i], mx);
            float al = __expf(m_i[i] - nm);
            float p0 = __expf(v0 - nm);
            float p1 = __expf(v1 - nm);
            float p2 = __expf(v2 - nm);
            float p3 = __expf(v3 - nm);
            float rs = (p0 + p1) + (p2 + p3);
#pragma unroll
            for (int d = 1; d < 16; d <<= 1) rs += __shfl_xor(rs, d, 32);
            l_i[i] = l_i[i] * al + rs;
            m_i[i] = nm;
#pragma unroll
            for (int j2 = 0; j2 < 4; ++j2) acc[j2][i] *= al;
            int r = i + half * 8;
            sP[wave][r][lc]      = (bf16_t)p0;
            sP[wave][r][16 + lc] = (bf16_t)p1;
            sP[wave][r][32 + lc] = (bf16_t)p2;
            sP[wave][r][48 + lc] = (bf16_t)p3;
        }
        // Re-read probs in A-fragment layout (wave-private LDS, DS in-order)
        v16bf pa0 = load_a_frag(&sP[wave][lc][0], half);
        v16bf pa1 = load_a_frag(&sP[wave][lc][32], half);
#pragma unroll
        for (int j2 = 0; j2 < 4; ++j2) {
            v16bf vb0 = load_b_frag(&sVt[j2 * 16 + lc][half * 16]);
            acc[j2] = __builtin_amdgcn_wmma_f32_16x16x32_bf16(
                false, pa0, false, vb0, (short)0, acc[j2], false, false);
            v16bf vb1 = load_b_frag(&sVt[j2 * 16 + lc][32 + half * 16]);
            acc[j2] = __builtin_amdgcn_wmma_f32_16x16x32_bf16(
                false, pa1, false, vb1, (short)0, acc[j2], false, false);
        }
        __syncthreads();   // protect sK/sVt before next stage
    }

    // epilogue: normalize and scatter to attn buffer [B*L, CA]
#pragma unroll
    for (int j2 = 0; j2 < 4; ++j2)
#pragma unroll
        for (int i = 0; i < 8; ++i) {
            int l = l0 + i + half * 8;
            float o = acc[j2][i] / l_i[i];
            attnb[((size_t)(b * LSEQ + l)) * CAD + head * DHEAD + j2 * 16 + lc] = (bf16_t)o;
        }
}

// ---------------------------------------------------------------------------
// Launch
// ---------------------------------------------------------------------------
extern "C" void kernel_launch(void* const* d_in, const int* in_sizes, int n_in,
                              void* d_out, int out_size, void* d_ws, size_t ws_size,
                              hipStream_t stream) {
    const float* hs    = (const float*)d_in[0];  // [2,1024,1024]
    const float* cah   = (const float*)d_in[1];  // [2,4096,1024]
    const float* mask  = (const float*)d_in[2];  // [2,4096]
    const float* scale = (const float*)d_in[3];  // [1024]
    const float* Wq    = (const float*)d_in[4];  // [1024,1024]
    const float* Wkv   = (const float*)d_in[5];  // [1024,2048]
    const float* Wo    = (const float*)d_in[6];  // [1024,1024]
    float* out = (float*)d_out;

    bf16_t* ws = (bf16_t*)d_ws;
    size_t off = 0;
    bf16_t* xb    = ws + off; off += (size_t)2048 * 1024;   // rmsnorm(x)
    bf16_t* cab   = ws + off; off += (size_t)8192 * 1024;   // ca bf16
    bf16_t* Wq_t  = ws + off; off += (size_t)1024 * 1024;   // Wq^T
    bf16_t* Wkv_t = ws + off; off += (size_t)2048 * 1024;   // Wkv^T
    bf16_t* Wo_t  = ws + off; off += (size_t)1024 * 1024;   // Wo^T
    bf16_t* qbuf  = ws + off; off += (size_t)2048 * 1024;   // q  [B*L, CA]
    bf16_t* kvbuf = ws + off; off += (size_t)8192 * 2048;   // kv [B*S, 2CA]
    bf16_t* attn  = ws + off; off += (size_t)2048 * 1024;   // attn out [B*L, CA]

    rmsnorm_kernel<<<2048, 256, 0, stream>>>(hs, scale, xb);
    cvt_bf16_kernel<<<(8192 * 1024 + 255) / 256, 256, 0, stream>>>(cah, cab, 8192 * 1024);
    transpose_cvt_kernel<<<dim3(1024 / 32, 1024 / 32), 256, 0, stream>>>(Wq, Wq_t, 1024, 1024);
    transpose_cvt_kernel<<<dim3(2048 / 32, 1024 / 32), 256, 0, stream>>>(Wkv, Wkv_t, 1024, 2048);
    transpose_cvt_kernel<<<dim3(1024 / 32, 1024 / 32), 256, 0, stream>>>(Wo, Wo_t, 1024, 1024);

    {   // q = x @ Wq  (M=2048, N=1024, K=1024)
        int tiles = (2048 / 32) * (1024 / 64);
        gemm_wmma_kernel<false><<<(tiles + 7) / 8, 256, 0, stream>>>(
            xb, Wq_t, qbuf, nullptr, 2048, 1024, 1024);
    }
    {   // kv = ca @ Wkv  (M=8192, N=2048, K=1024)
        int tiles = (8192 / 32) * (2048 / 64);
        gemm_wmma_kernel<false><<<(tiles + 7) / 8, 256, 0, stream>>>(
            cab, Wkv_t, kvbuf, nullptr, 8192, 2048, 1024);
    }
    attn_kernel<<<dim3(BATCH * NHEADS, LSEQ / 128), 256, 0, stream>>>(qbuf, kvbuf, mask, attn);
    {   // out = attn @ Wo  (f32 out)
        int tiles = (2048 / 32) * (1024 / 64);
        gemm_wmma_kernel<true><<<(tiles + 7) / 8, 256, 0, stream>>>(
            attn, Wo_t, nullptr, out, 2048, 1024, 1024);
    }
}
